// SNN_MLP_STDP_45337674776885
// MI455X (gfx1250) — matrix operand order repaired and checked by reading, not compiled, not run on hardware
//
#include <hip/hip_runtime.h>
#include <hip/hip_bf16.h>

// ---------------- problem constants ----------------
#define T_STEPS 100
#define BATCH   256
#define IN_DIM  784
#define KP      800                      // IN_DIM zero-padded to multiple of 32
#define HID_DIM 1024
#define OUT_DIM 10
#define M1      (T_STEPS * BATCH)        // 25600
#define THRESH  0.9f
#define DECAY   0.6f

typedef __attribute__((ext_vector_type(16))) _Float16 v16h;
typedef __attribute__((ext_vector_type(8)))  _Float16 v8h;
typedef __attribute__((ext_vector_type(8)))  float    v8f;

union Frag16 { v16h v; v8h h[2]; };

// ---------------- CDNA5 async global->LDS copy (ASYNCcnt path) ----------------
#if __has_builtin(__builtin_amdgcn_global_load_async_to_lds_b128)
#define HAVE_ASYNC_LDS 1
#else
#define HAVE_ASYNC_LDS 0
#endif

// builtin expects GCC-vector int4 pointers with explicit address spaces
typedef int v4i_vs __attribute__((vector_size(16)));
typedef v4i_vs __attribute__((address_space(1)))* gbl_v4i_p;
typedef v4i_vs __attribute__((address_space(3)))* lds_v4i_p;

__device__ __forceinline__ void cp16_g2l(const _Float16* g, _Float16* l) {
#if HAVE_ASYNC_LDS
    __builtin_amdgcn_global_load_async_to_lds_b128(
        (gbl_v4i_p)(void*)g, (lds_v4i_p)(void*)l, 0, 0);
#else
    *(v8h*)l = *(const v8h*)g;
#endif
}

__device__ __forceinline__ void wait_async_lds() {
#if HAVE_ASYNC_LDS
 #if __has_builtin(__builtin_amdgcn_s_wait_asynccnt)
    __builtin_amdgcn_s_wait_asynccnt(0);
 #else
    asm volatile("s_wait_asynccnt 0" ::: "memory");
 #endif
#endif
}

// ---------------- padding / precision conversions ----------------
// Xh[m][k] = (k<784) ? f16(X[m][k]) : 0        m<25600, k<800
__global__ void cvt_x_pad(const float* __restrict__ X, _Float16* __restrict__ Xh) {
    long i = (long)blockIdx.x * 256 + threadIdx.x;
    if (i < (long)M1 * KP) {
        int m = (int)(i / KP), k = (int)(i % KP);
        Xh[i] = (k < IN_DIM) ? (_Float16)X[(size_t)m * IN_DIM + k] : (_Float16)0.f;
    }
}
// W1t[n][k] = (k<784) ? f16(W1[k][n]) : 0      (transposed: N-major, K padded)
__global__ void cvt_w1_tr(const float* __restrict__ W1, _Float16* __restrict__ W1t) {
    int i = blockIdx.x * 256 + threadIdx.x;
    if (i < HID_DIM * KP) {
        int n = i / KP, k = i % KP;
        W1t[i] = (k < IN_DIM) ? (_Float16)W1[(size_t)k * HID_DIM + n] : (_Float16)0.f;
    }
}
// W2t[n][k] = W2[k][n] for n<10 else 0         (N-major, padded to 16 rows)
__global__ void cvt_w2_tr(const float* __restrict__ W2, _Float16* __restrict__ W2t) {
    int i = blockIdx.x * 256 + threadIdx.x;
    if (i < 16 * HID_DIM) {
        int n = i >> 10, k = i & 1023;
        W2t[i] = (n < OUT_DIM) ? (_Float16)W2[k * OUT_DIM + n] : (_Float16)0.f;
    }
}

// ---------------- GEMM1: U1[M1,1024] = Xh[M1,800] @ W1t^T ----------------
// 256 threads = 8 waves (4M x 2N), tile 128x128x32, double-buffered LDS,
// async global->LDS staging overlapped with 8 wmma_f32_16x16x32_f16 per wave.
#define NKB (KP / 32)    // 25
#define LDP 40           // LDS row pitch (halves): conflict-free b128 reads

__global__ __launch_bounds__(256) void gemm1_wmma(
        const _Float16* __restrict__ Xh,   // [M1, 800]
        const _Float16* __restrict__ W1t,  // [1024, 800]
        float*          __restrict__ U1)   // [M1, 1024] (= d_out hidden region)
{
    __shared__ _Float16 As[2][128][LDP];   // M x K, row-major
    __shared__ _Float16 Bs[2][128][LDP];   // N x K (B pre-transposed in global)

    const int tid   = threadIdx.x;
    const int lane  = tid & 31;
    const int wave  = tid >> 5;
    const int waveM = wave >> 1;           // 0..3
    const int waveN = wave & 1;            // 0..1
    const int m0    = blockIdx.x * 128;
    const int n0    = blockIdx.y * 128;
    const int half  = lane >> 4;
    const int l16   = lane & 15;

    // staging decomposition: 128 rows x 32 halves = 256 threads x 2 x 16B
    const int srow = tid >> 2;             // 0..63
    const int sk8  = (tid & 3) * 8;        // 0,8,16,24

    const _Float16* aSrc = Xh  + (size_t)(m0 + srow) * KP + sk8;
    const _Float16* bSrc = W1t + (size_t)(n0 + srow) * KP + sk8;

    v8f acc[2][4] = {};

    // prologue: stage k-block 0 into buffer 0
    cp16_g2l(aSrc,                 &As[0][srow][sk8]);
    cp16_g2l(aSrc + (size_t)64*KP, &As[0][srow + 64][sk8]);
    cp16_g2l(bSrc,                 &Bs[0][srow][sk8]);
    cp16_g2l(bSrc + (size_t)64*KP, &Bs[0][srow + 64][sk8]);
    wait_async_lds();
    __syncthreads();

    for (int kb = 0; kb < NKB; ++kb) {
        const int cur = kb & 1;
        // ---- overlap: async-stage k-block kb+1 into the other buffer ----
        if (kb + 1 < NKB) {
            const int nxt = cur ^ 1;
            const int go  = (kb + 1) * 32;
            cp16_g2l(aSrc + go,                 &As[nxt][srow][sk8]);
            cp16_g2l(aSrc + go + (size_t)64*KP, &As[nxt][srow + 64][sk8]);
            cp16_g2l(bSrc + go,                 &Bs[nxt][srow][sk8]);
            cp16_g2l(bSrc + go + (size_t)64*KP, &Bs[nxt][srow + 64][sk8]);
        }

        // ---- fragments + WMMA on current buffer ----
        Frag16 af[2], bf[4];
        #pragma unroll
        for (int mt = 0; mt < 2; ++mt) {
            const int r = waveM * 32 + mt * 16 + l16;
            af[mt].h[0] = *(const v8h*)&As[cur][r][8 * half];
            af[mt].h[1] = *(const v8h*)&As[cur][r][16 + 8 * half];
        }
        #pragma unroll
        for (int nt = 0; nt < 4; ++nt) {
            const int n = waveN * 64 + nt * 16 + l16;
            bf[nt].h[0] = *(const v8h*)&Bs[cur][n][8 * half];
            bf[nt].h[1] = *(const v8h*)&Bs[cur][n][16 + 8 * half];
        }
        #pragma unroll
        for (int mt = 0; mt < 2; ++mt)
            #pragma unroll
            for (int nt = 0; nt < 4; ++nt)
                acc[mt][nt] = __builtin_amdgcn_wmma_f32_16x16x32_f16(
                    false, af[mt].v, false, bf[nt].v,
                    (short)0, acc[mt][nt], false, false);

        wait_async_lds();
        __syncthreads();
    }

    // ---- epilogue: C/D layout VGPR j -> M = j + 8*half, N = lane&15 ----
    #pragma unroll
    for (int mt = 0; mt < 2; ++mt) {
        const int mrow = m0 + waveM * 32 + mt * 16 + 8 * half;
        #pragma unroll
        for (int nt = 0; nt < 4; ++nt) {
            const int n = n0 + waveN * 64 + nt * 16 + l16;
            #pragma unroll
            for (int j = 0; j < 8; ++j)
                U1[(size_t)(mrow + j) * HID_DIM + n] = acc[mt][nt][j];
        }
    }
}

// ---------------- LIF1: in-place scan over T, emit f32 spikes + f16 copy ----
__global__ __launch_bounds__(256) void lif1_scan(
        float* __restrict__ U, _Float16* __restrict__ S1h) {
    const int idx = blockIdx.x * 256 + threadIdx.x;     // 0..B*H-1
    const size_t stride = (size_t)BATCH * HID_DIM;
    float mem = 0.f;
    float*    p = U + idx;
    _Float16* q = S1h + idx;
    for (int t = 0; t < T_STEPS; ++t) {
        const size_t off = (size_t)t * stride;
        float u = p[off];
        mem = mem * DECAY + u;
        float s = (mem >= THRESH) ? 1.f : 0.f;
        mem -= s * THRESH;
        p[off] = s;
        q[off] = (_Float16)s;
    }
}

// ---------------- GEMM2: out2[M1,10] = S1h[M1,1024] @ W2t^T ----------------
__global__ __launch_bounds__(256) void gemm2_wmma(
        const _Float16* __restrict__ S1h,   // [M1, 1024]
        const _Float16* __restrict__ W2t,   // [16, 1024] (N-major)
        float*          __restrict__ out2)  // [M1, 10]
{
    const int lane = threadIdx.x & 31;
    const int wave = threadIdx.x >> 5;
    const int row0 = (blockIdx.x * 8 + wave) * 16;
    const int half = lane >> 4;
    const int l16  = lane & 15;

    const _Float16* aBase = S1h + (size_t)(row0 + l16) * HID_DIM;
    const _Float16* bBase = W2t + (size_t)l16 * HID_DIM;

    v8f acc = {};
    #pragma unroll 4
    for (int kb = 0; kb < HID_DIM / 32; ++kb) {
        const int k0 = kb * 32;
        Frag16 a, b;
        a.h[0] = *(const v8h*)(aBase + k0 + 8 * half);
        a.h[1] = *(const v8h*)(aBase + k0 + 16 + 8 * half);
        b.h[0] = *(const v8h*)(bBase + k0 + 8 * half);
        b.h[1] = *(const v8h*)(bBase + k0 + 16 + 8 * half);
        acc = __builtin_amdgcn_wmma_f32_16x16x32_f16(
                false, a.v, false, b.v, (short)0, acc, false, false);
    }

    if (l16 < OUT_DIM) {
        #pragma unroll
        for (int j = 0; j < 8; ++j) {
            const int m = row0 + j + 8 * half;
            out2[(size_t)m * OUT_DIM + l16] = acc[j];
        }
    }
}

// ---------------- LIF2: in-place scan over T on output region ----------------
__global__ __launch_bounds__(256) void lif2_scan(float* __restrict__ U2) {
    const int idx = blockIdx.x * 256 + threadIdx.x;     // 0..B*O-1
    const size_t stride = (size_t)BATCH * OUT_DIM;
    float mem = 0.f;
    float* p = U2 + idx;
    for (int t = 0; t < T_STEPS; ++t) {
        const size_t off = (size_t)t * stride;
        float u = p[off];
        mem = mem * DECAY + u;
        float s = (mem >= THRESH) ? 1.f : 0.f;
        mem -= s * THRESH;
        p[off] = s;
    }
}

// ---------------- launcher ----------------
extern "C" void kernel_launch(void* const* d_in, const int* in_sizes, int n_in,
                              void* d_out, int out_size, void* d_ws, size_t ws_size,
                              hipStream_t stream) {
    const float* X  = (const float*)d_in[0];   // [T,B,784]
    const float* W1 = (const float*)d_in[1];   // [784,1024]
    const float* W2 = (const float*)d_in[2];   // [1024,10]

    float* out  = (float*)d_out;
    float* U1   = out;                              // hidden region doubles as U1
    float* out2 = out + (size_t)M1 * HID_DIM;       // output region

    // workspace layout (all 256B-aligned)
    char* ws = (char*)d_ws;
    const size_t S1H_BYTES = (size_t)M1 * HID_DIM * 2;      // 52,428,800
    const size_t XH_BYTES  = (size_t)M1 * KP * 2;           // 40,960,000
    const size_t W1T_BYTES = (size_t)HID_DIM * KP * 2;      //  1,638,400
    _Float16* S1h = (_Float16*)ws;
    _Float16* Xh  = (_Float16*)(ws + S1H_BYTES);
    _Float16* W1t = (_Float16*)(ws + S1H_BYTES + XH_BYTES);
    _Float16* W2t = (_Float16*)(ws + S1H_BYTES + XH_BYTES + W1T_BYTES);

    cvt_x_pad <<<(int)(((long)M1 * KP + 255) / 256), 256, 0, stream>>>(X, Xh);
    cvt_w1_tr <<<(HID_DIM * KP + 255) / 256, 256, 0, stream>>>(W1, W1t);
    cvt_w2_tr <<<(16 * HID_DIM + 255) / 256, 256, 0, stream>>>(W2, W2t);

    gemm1_wmma<<<dim3(M1 / 128, HID_DIM / 128), 256, 0, stream>>>(Xh, W1t, U1);
    lif1_scan <<<(BATCH * HID_DIM) / 256, 256, 0, stream>>>(U1, S1h);
    gemm2_wmma<<<M1 / 128, 256, 0, stream>>>(S1h, W2t, out2);
    lif2_scan <<<(BATCH * OUT_DIM) / 256, 256, 0, stream>>>(out2);
}
